// CausalGraphVAE_51470888075722
// MI455X (gfx1250) — matrix operand order repaired and checked by reading, not compiled, not run on hardware
//
#include <hip/hip_runtime.h>
#include <cstdint>
#include <cstddef>

// ---------------------------------------------------------------------------
// CausalGraphVAE forward for MI455X (gfx1250, wave32, WMMA + async-LDS).
//
// H0 == 0 in every TGCN cell => R gate dead, cell = (1-Z)*Ht.
// N x N aggregation GEMMs: bf16 WMMA, hi/lo split precision
//   out = Ahi*Mhi + Ahi*Mlo + Alo*Mhi   (~2^-17 rel. error, fp32-grade).
// GEMM kernel: 4 waves/block, 64x64 block tile, 16x64 per wave.
//   B tiles (64x32, hi+lo) staged in LDS via GLOBAL_LOAD_ASYNC_TO_LDS_B128,
//   double buffered (s_wait_asynccnt + barrier), A streamed from global
//   (L2-resident, 192MB L2 >> 64MB AnT) with global_prefetch ahead.
//
// d_in order (setup_inputs() dict insertion order, recursive):
//  0 x 1 entity_emb 2 time_emb 3 num_nodes 4 edge_score
//  5 W_ent 6 b_ent 7 W_time 8 b_time 9 att1
//  10..13 t1.z(Wc,bc,Wl,bl) 14..17 t1.r(unused) 18..21 t1.h
//  22 W_mu 23 b_mu 24 W_lv 25 b_lv 26 W_dec 27 b_dec
//  28..31 td.z 32..35 td.r(unused) 36..39 td.h
//
// d_out = [recon_x(4096*64) | mu(4096*64) | logvar(4096*64) | A(4096*4096)]
// ---------------------------------------------------------------------------

#define NN 4096

typedef __attribute__((ext_vector_type(16))) __bf16 v16bf;
typedef __attribute__((ext_vector_type(8)))  __bf16 v8bf;
typedef __attribute__((ext_vector_type(8)))  float  v8f;

// ---- bf16 helpers (manual RNE, storage as ushort) -------------------------
__device__ __forceinline__ unsigned short f2bf_rne(float f) {
  unsigned u = __float_as_uint(f);
  unsigned rnd = 0x7fffu + ((u >> 16) & 1u);
  return (unsigned short)((u + rnd) >> 16);
}
__device__ __forceinline__ float bf2f(unsigned short h) {
  return __uint_as_float(((unsigned)h) << 16);
}
__device__ __forceinline__ void store_split(float v, unsigned short* ph, unsigned short* pl) {
  unsigned short h = f2bf_rne(v);
  *ph = h;
  *pl = f2bf_rne(v - bf2f(h));
}

// ---- jax.random.normal(key(42), (4096,64)) : threefry2x32 + erfinv --------
__device__ __forceinline__ unsigned rotl32(unsigned v, int n) {
  return (v << n) | (v >> (32 - n));
}
__device__ float jax_normal_42(unsigned idx) {
  const unsigned half = (4096u * 64u) / 2u;  // 131072
  unsigned i = (idx < half) ? idx : (idx - half);
  unsigned x0 = i, x1 = i + half;
  const unsigned k0 = 0u, k1 = 42u;
  const unsigned k2 = k0 ^ k1 ^ 0x1BD11BDAu;
  x0 += k0; x1 += k1;
#define TFR(r) { x0 += x1; x1 = rotl32(x1, r); x1 ^= x0; }
  TFR(13) TFR(15) TFR(26) TFR(6)  x0 += k1; x1 += k2 + 1u;
  TFR(17) TFR(29) TFR(16) TFR(24) x0 += k2; x1 += k0 + 2u;
  TFR(13) TFR(15) TFR(26) TFR(6)  x0 += k0; x1 += k1 + 3u;
  TFR(17) TFR(29) TFR(16) TFR(24) x0 += k1; x1 += k2 + 4u;
  TFR(13) TFR(15) TFR(26) TFR(6)  x0 += k2; x1 += k0 + 5u;
#undef TFR
  unsigned bits = (idx < half) ? x0 : x1;
  float u01 = __uint_as_float((bits >> 9) | 0x3f800000u) - 1.0f;  // [0,1)
  const float lo = -0.99999994f;                                  // nextafter(-1,0)
  float u = fmaf(u01, 1.0f - lo, lo);
  u = fmaxf(u, lo);
  return 1.41421356237f * erfinvf(u);
}

// ---- K1: A = sigmoid(edge_score) (written straight to the A output) -------
__global__ __launch_bounds__(256)
void k_sigA(const float4* __restrict__ es, float4* __restrict__ A) {
  int i = blockIdx.x * 256 + threadIdx.x;   // NN*NN/4 elements
  float4 v = es[i];
  float4 r;
  r.x = 1.0f / (1.0f + expf(-v.x));
  r.y = 1.0f / (1.0f + expf(-v.y));
  r.z = 1.0f / (1.0f + expf(-v.z));
  r.w = 1.0f / (1.0f + expf(-v.w));
  A[i] = r;
}

// ---- K2a/K2b: dis[i] = rsqrt(sum_j A[j,i]) two-stage reduction ------------
__global__ __launch_bounds__(256)
void k_degpart(const float* __restrict__ A, float* __restrict__ part) {
  int col = blockIdx.x * 256 + threadIdx.x;
  int rc = blockIdx.y;                       // 16 row chunks of 256 rows
  float s = 0.0f;
  for (int j = rc * 256; j < rc * 256 + 256; ++j) s += A[(size_t)j * NN + col];
  part[(size_t)rc * NN + col] = s;
}
__global__ __launch_bounds__(256)
void k_degfin(const float* __restrict__ part, float* __restrict__ dis) {
  int col = blockIdx.x * 256 + threadIdx.x;
  float s = 0.0f;
  for (int r = 0; r < 16; ++r) s += part[(size_t)r * NN + col];
  dis[col] = rsqrtf(s);
}

// ---- K3: AnT[i,j] = A[j,i]*dis[i]*dis[j], split into bf16 hi/lo -----------
__global__ __launch_bounds__(256)
void k_make_AnT(const float* __restrict__ A, const float* __restrict__ dis,
                unsigned short* __restrict__ Ah, unsigned short* __restrict__ Al) {
  __shared__ float tile[32][33];
  int bi = blockIdx.x, bj = blockIdx.y;           // out-row / out-col 32-blocks
  int tx = threadIdx.x, ty = threadIdx.y;         // 32 x 8
#pragma unroll
  for (int l = 0; l < 4; ++l) {
    int j = bj * 32 + ty + l * 8;                 // A row (source)
    tile[ty + l * 8][tx] = A[(size_t)j * NN + bi * 32 + tx];
  }
  __syncthreads();
#pragma unroll
  for (int l = 0; l < 4; ++l) {
    int i = bi * 32 + ty + l * 8;                 // AnT row (target)
    int j = bj * 32 + tx;                         // AnT col (source)
    float v = tile[tx][ty + l * 8] * dis[i] * dis[j];
    size_t o = (size_t)i * NN + j;
    store_split(v, Ah + o, Al + o);
  }
}

// ---- K4: encoder features -> gate matrices M, stored transposed bf16 ------
// col layout of M: (p*2 + g)*128 + c, g: 0=z 1=h.  MT[col][node].
__global__ __launch_bounds__(128)
void k_enc_feat(const float* __restrict__ x, const float* __restrict__ eemb,
                const float* __restrict__ temb,
                const float* __restrict__ W_ent, const float* __restrict__ b_ent,
                const float* __restrict__ W_time, const float* __restrict__ b_time,
                const float* __restrict__ Wc_z, const float* __restrict__ Wc_h,
                unsigned short* __restrict__ MTh, unsigned short* __restrict__ MTl) {
  int node = blockIdx.x, p = blockIdx.y, c = threadIdx.x;
  __shared__ float xc[320];
  __shared__ float ebuf[128];
  size_t base = (size_t)node * 3 + p;
  if (c < 64) xc[c] = x[base * 64 + c];
  ebuf[c] = eemb[base * 128 + c];
  __syncthreads();
  float acc = b_ent[c];
  for (int k = 0; k < 128; ++k) acc = fmaf(ebuf[k], W_ent[k * 128 + c], acc);
  __syncthreads();
  xc[64 + c] = fmaxf(acc, 0.0f);
  ebuf[c] = temb[base * 128 + c];
  __syncthreads();
  acc = b_time[c];
  for (int k = 0; k < 128; ++k) acc = fmaf(ebuf[k], W_time[k * 128 + c], acc);
  xc[192 + c] = fmaxf(acc, 0.0f);
  __syncthreads();
#pragma unroll
  for (int g = 0; g < 2; ++g) {
    const float* W = g ? Wc_h : Wc_z;             // (320 x 128)
    float a = 0.0f;
    for (int k = 0; k < 320; ++k) a = fmaf(xc[k], W[k * 128 + c], a);
    size_t o = (size_t)((p * 2 + g) * 128 + c) * NN + node;
    store_split(a, MTh + o, MTl + o);
  }
}

// ---- K5: split-precision bf16 WMMA GEMM with async-LDS B staging ----------
// One wave -> 16x64 tile; block = 4 waves -> 64x64. B tile double-buffered in
// LDS via GLOBAL_LOAD_ASYNC_TO_LDS_B128 (ASYNCcnt), shared by all 4 waves.
__global__ __launch_bounds__(128)
void k_gemm_split(const unsigned short* __restrict__ Ah,
                  const unsigned short* __restrict__ Al,
                  const unsigned short* __restrict__ Bh,
                  const unsigned short* __restrict__ Bl,
                  float* __restrict__ out, int outc) {
  // [buf][hi/lo][col_local 64][k_local 32] bf16 -> 16 KB
  __shared__ __align__(64) unsigned short Bsh[2][2][64][32];

  int tid = threadIdx.x;
  int lane = tid & 31;
  int wave = tid >> 5;
  int rowTile = blockIdx.y * 64 + wave * 16;
  int colTile = blockIdx.x * 64;
  int row = rowTile + (lane & 15);
  const size_t rowOff = (size_t)row * NN;
  const int aoff = (lane & 16) ? 16 : 0;
  const int bkoff = (lane & 16) ? 16 : 0;

  unsigned ldsBase = (unsigned)(unsigned long long)(void*)&Bsh[0][0][0][0];

  auto issue = [&](int k, int b) {
    // copy 64 cols x 32 k x bf16 (hi and lo) = 2 x 4 KB, 16B per async op
#pragma unroll
    for (int it = 0; it < 2; ++it) {
      int t = tid + it * 128;                 // 0..255
      int col = t >> 2, part = t & 3;
      size_t goff = (((size_t)(colTile + col)) * NN + k) * 2 + (size_t)part * 16;
      unsigned l0 = ldsBase + (unsigned)((b * 2 + 0) * 4096 + col * 64 + part * 16);
      unsigned l1 = ldsBase + (unsigned)((b * 2 + 1) * 4096 + col * 64 + part * 16);
      const char* g0 = (const char*)Bh + goff;
      const char* g1 = (const char*)Bl + goff;
      asm volatile("global_load_async_to_lds_b128 %0, %1, off"
                   :: "v"(l0), "v"(g0) : "memory");
      asm volatile("global_load_async_to_lds_b128 %0, %1, off"
                   :: "v"(l1), "v"(g1) : "memory");
    }
  };

  v8f acc[4] = {v8f{}, v8f{}, v8f{}, v8f{}};
  issue(0, 0);
  int buf = 0;
  for (int ks = 0; ks < NN / 32; ++ks) {
    int k = ks * 32;
    asm volatile("s_wait_asynccnt 0" ::: "memory");
    __syncthreads();
    if (ks + 1 < NN / 32) issue(k + 32, buf ^ 1);

    // A fragments (hi/lo) from global; prefetch the stream ahead.
    const unsigned short* pa = Ah + rowOff + k + aoff;
    const unsigned short* pl = Al + rowOff + k + aoff;
    v16bf ah, al;
    ((v8bf*)&ah)[0] = *(const v8bf*)pa;
    ((v8bf*)&ah)[1] = *(const v8bf*)(pa + 8);
    ((v8bf*)&al)[0] = *(const v8bf*)pl;
    ((v8bf*)&al)[1] = *(const v8bf*)(pl + 8);
    __builtin_prefetch(pa + 256, 0, 3);
    __builtin_prefetch(pl + 256, 0, 3);

#pragma unroll
    for (int cc = 0; cc < 4; ++cc) {
      int cl = cc * 16 + (lane & 15);
      v16bf bh = *(const v16bf*)&Bsh[buf][0][cl][bkoff];
      v16bf bl = *(const v16bf*)&Bsh[buf][1][cl][bkoff];
      acc[cc] = __builtin_amdgcn_wmma_f32_16x16x32_bf16(
          false, ah, false, bh, (short)0, acc[cc], false, false);
      acc[cc] = __builtin_amdgcn_wmma_f32_16x16x32_bf16(
          false, al, false, bh, (short)0, acc[cc], false, false);
      acc[cc] = __builtin_amdgcn_wmma_f32_16x16x32_bf16(
          false, ah, false, bl, (short)0, acc[cc], false, false);
    }
    buf ^= 1;
  }

  int n = lane & 15;
  int mAdd = (lane >> 4) * 8;
#pragma unroll
  for (int cc = 0; cc < 4; ++cc)
#pragma unroll
    for (int r = 0; r < 8; ++r)
      out[(size_t)(rowTile + mAdd + r) * outc + colTile + cc * 16 + n] = acc[cc][r];
}

// ---- K6: encoder gates + attention sum + relu -> h ------------------------
__global__ __launch_bounds__(128)
void k_enc_gate(const float* __restrict__ G1,
                const float* __restrict__ bc_z, const float* __restrict__ bc_h,
                const float* __restrict__ Wl_z, const float* __restrict__ bl_z,
                const float* __restrict__ Wl_h, const float* __restrict__ bl_h,
                const float* __restrict__ att, float* __restrict__ hout) {
  int node = blockIdx.x, c = threadIdx.x;
  __shared__ float gz[128], gh[128];
  float a0 = att[0], a1 = att[1], a2 = att[2];
  float m = fmaxf(a0, fmaxf(a1, a2));
  float e0 = expf(a0 - m), e1 = expf(a1 - m), e2 = expf(a2 - m);
  float inv = 1.0f / (e0 + e1 + e2);
  float probs[3] = {e0 * inv, e1 * inv, e2 * inv};
  float henc = 0.0f;
  for (int p = 0; p < 3; ++p) {
    gz[c] = G1[(size_t)node * 768 + (p * 2 + 0) * 128 + c] + bc_z[c];
    gh[c] = G1[(size_t)node * 768 + (p * 2 + 1) * 128 + c] + bc_h[c];
    __syncthreads();
    float az = bl_z[c], ah = bl_h[c];
    for (int k = 0; k < 128; ++k) {
      az = fmaf(gz[k], Wl_z[k * 128 + c], az);
      ah = fmaf(gh[k], Wl_h[k * 128 + c], ah);
    }
    float Z = 1.0f / (1.0f + expf(-az));
    float Ht = tanhf(ah);
    henc = fmaf(probs[p], (1.0f - Z) * Ht, henc);
    __syncthreads();
  }
  hout[(size_t)node * 128 + c] = fmaxf(henc, 0.0f);
}

// ---- K7: VAE head: mu, logvar, z = mu + eps*exp(lv/2), d = z@W_dec + b ----
__global__ __launch_bounds__(128)
void k_vae(const float* __restrict__ h,
           const float* __restrict__ W_mu, const float* __restrict__ b_mu,
           const float* __restrict__ W_lv, const float* __restrict__ b_lv,
           const float* __restrict__ W_dec, const float* __restrict__ b_dec,
           float* __restrict__ mu_out, float* __restrict__ lv_out,
           float* __restrict__ dvec) {
  int node = blockIdx.x, c = threadIdx.x;
  __shared__ float hb[128], zb[64];
  hb[c] = h[(size_t)node * 128 + c];
  __syncthreads();
  if (c < 64) {
    float mu = b_mu[c], lv = b_lv[c];
    for (int k = 0; k < 128; ++k) {
      mu = fmaf(hb[k], W_mu[k * 64 + c], mu);
      lv = fmaf(hb[k], W_lv[k * 64 + c], lv);
    }
    unsigned idx = (unsigned)node * 64u + (unsigned)c;
    float eps = jax_normal_42(idx);
    mu_out[idx] = mu;
    lv_out[idx] = lv;
    zb[c] = fmaf(eps, expf(0.5f * lv), mu);
  }
  __syncthreads();
  float d = b_dec[c];
  for (int k = 0; k < 64; ++k) d = fmaf(zb[k], W_dec[k * 128 + c], d);
  dvec[(size_t)node * 128 + c] = d;
}

// ---- K8: decoder features -> M2 transposed bf16 (cols: [z:0..63 | h:64..127])
__global__ __launch_bounds__(128)
void k_dec_feat(const float* __restrict__ dvec,
                const float* __restrict__ Wc_z, const float* __restrict__ Wc_h,
                unsigned short* __restrict__ MTh, unsigned short* __restrict__ MTl) {
  int node = blockIdx.x, c = threadIdx.x;
  __shared__ float db[128];
  db[c] = dvec[(size_t)node * 128 + c];
  __syncthreads();
  const float* W = (c >= 64) ? Wc_h : Wc_z;   // (128 x 64)
  int cc = c & 63;
  float a = 0.0f;
  for (int k = 0; k < 128; ++k) a = fmaf(db[k], W[k * 64 + cc], a);
  size_t o = (size_t)c * NN + node;
  store_split(a, MTh + o, MTl + o);
}

// ---- K10: decoder gate -> recon_x -----------------------------------------
__global__ __launch_bounds__(64)
void k_dec_gate(const float* __restrict__ G2,
                const float* __restrict__ bc_z, const float* __restrict__ bc_h,
                const float* __restrict__ Wl_z, const float* __restrict__ bl_z,
                const float* __restrict__ Wl_h, const float* __restrict__ bl_h,
                float* __restrict__ recon) {
  int node = blockIdx.x, c = threadIdx.x;
  __shared__ float gz[64], gh[64];
  gz[c] = G2[(size_t)node * 128 + c] + bc_z[c];
  gh[c] = G2[(size_t)node * 128 + 64 + c] + bc_h[c];
  __syncthreads();
  float az = bl_z[c], ah = bl_h[c];
  for (int k = 0; k < 64; ++k) {
    az = fmaf(gz[k], Wl_z[k * 64 + c], az);
    ah = fmaf(gh[k], Wl_h[k * 64 + c], ah);
  }
  float Z = 1.0f / (1.0f + expf(-az));
  float Ht = tanhf(ah);
  recon[(size_t)node * 64 + c] = fmaxf((1.0f - Z) * Ht, 0.0f);
}

// ---------------------------------------------------------------------------
extern "C" void kernel_launch(void* const* d_in, const int* in_sizes, int n_in,
                              void* d_out, int out_size, void* d_ws, size_t ws_size,
                              hipStream_t stream) {
  (void)in_sizes; (void)n_in; (void)out_size; (void)ws_size;

  const float* x      = (const float*)d_in[0];
  const float* eemb   = (const float*)d_in[1];
  const float* temb   = (const float*)d_in[2];
  const float* edge   = (const float*)d_in[4];
  const float* W_ent  = (const float*)d_in[5];
  const float* b_ent  = (const float*)d_in[6];
  const float* W_time = (const float*)d_in[7];
  const float* b_time = (const float*)d_in[8];
  const float* att1   = (const float*)d_in[9];
  const float* t1_Wc_z = (const float*)d_in[10];
  const float* t1_bc_z = (const float*)d_in[11];
  const float* t1_Wl_z = (const float*)d_in[12];
  const float* t1_bl_z = (const float*)d_in[13];
  const float* t1_Wc_h = (const float*)d_in[18];
  const float* t1_bc_h = (const float*)d_in[19];
  const float* t1_Wl_h = (const float*)d_in[20];
  const float* t1_bl_h = (const float*)d_in[21];
  const float* W_mu  = (const float*)d_in[22];
  const float* b_mu  = (const float*)d_in[23];
  const float* W_lv  = (const float*)d_in[24];
  const float* b_lv  = (const float*)d_in[25];
  const float* W_dec = (const float*)d_in[26];
  const float* b_dec = (const float*)d_in[27];
  const float* td_Wc_z = (const float*)d_in[28];
  const float* td_bc_z = (const float*)d_in[29];
  const float* td_Wl_z = (const float*)d_in[30];
  const float* td_bl_z = (const float*)d_in[31];
  const float* td_Wc_h = (const float*)d_in[36];
  const float* td_bc_h = (const float*)d_in[37];
  const float* td_Wl_h = (const float*)d_in[38];
  const float* td_bl_h = (const float*)d_in[39];

  float* recon = (float*)d_out;                 // 4096*64
  float* mu_o  = recon + 4096 * 64;             // 4096*64
  float* lv_o  = mu_o + 4096 * 64;              // 4096*64
  float* A_o   = lv_o + 4096 * 64;              // 4096*4096

  // workspace layout (bytes); ~96.3 MiB total
  char* ws = (char*)d_ws;
  const size_t OFF_DIS    = 0;
  const size_t OFF_AHI    = 65536;
  const size_t OFF_ALO    = OFF_AHI + (size_t)NN * NN * 2;       // +32 MB
  const size_t OFF_MET_HI = OFF_ALO + (size_t)NN * NN * 2;       // +32 MB
  const size_t OFF_MET_LO = OFF_MET_HI + (size_t)768 * NN * 2;   // +6 MB
  const size_t OFF_G1     = OFF_MET_LO + (size_t)768 * NN * 2;   // +6 MB
  const size_t OFF_H      = OFF_G1 + (size_t)NN * 768 * 4;       // +12.6 MB
  const size_t OFF_D      = OFF_H + (size_t)NN * 128 * 4;
  const size_t OFF_MDT_HI = OFF_D + (size_t)NN * 128 * 4;
  const size_t OFF_MDT_LO = OFF_MDT_HI + (size_t)128 * NN * 2;
  const size_t OFF_G2     = OFF_MDT_LO + (size_t)128 * NN * 2;
  const size_t OFF_PART   = OFF_G2 + (size_t)NN * 128 * 4;       // 16*NN floats

  float*          dis  = (float*)(ws + OFF_DIS);
  unsigned short* Ahi  = (unsigned short*)(ws + OFF_AHI);
  unsigned short* Alo  = (unsigned short*)(ws + OFF_ALO);
  unsigned short* METh = (unsigned short*)(ws + OFF_MET_HI);
  unsigned short* METl = (unsigned short*)(ws + OFF_MET_LO);
  float*          G1   = (float*)(ws + OFF_G1);
  float*          hbuf = (float*)(ws + OFF_H);
  float*          dvec = (float*)(ws + OFF_D);
  unsigned short* MDTh = (unsigned short*)(ws + OFF_MDT_HI);
  unsigned short* MDTl = (unsigned short*)(ws + OFF_MDT_LO);
  float*          G2   = (float*)(ws + OFF_G2);
  float*          part = (float*)(ws + OFF_PART);

  // 1. A = sigmoid(edge_score)
  k_sigA<<<(NN * NN / 4) / 256, 256, 0, stream>>>((const float4*)edge, (float4*)A_o);
  // 2. dis = rsqrt(colsum(A)), two-stage
  k_degpart<<<dim3(NN / 256, 16), 256, 0, stream>>>(A_o, part);
  k_degfin<<<NN / 256, 256, 0, stream>>>(part, dis);
  // 3. AnT -> bf16 hi/lo
  k_make_AnT<<<dim3(NN / 32, NN / 32), dim3(32, 8), 0, stream>>>(A_o, dis, Ahi, Alo);
  // 4. encoder features -> MencT hi/lo
  k_enc_feat<<<dim3(NN, 3), 128, 0, stream>>>(x, eemb, temb, W_ent, b_ent, W_time,
                                              b_time, t1_Wc_z, t1_Wc_h, METh, METl);
  // 5. G1 = AnT @ Menc  (split-precision bf16 WMMA, async-LDS B staging)
  k_gemm_split<<<dim3(768 / 64, NN / 64), 128, 0, stream>>>(Ahi, Alo, METh, METl, G1, 768);
  // 6. encoder gates -> h
  k_enc_gate<<<NN, 128, 0, stream>>>(G1, t1_bc_z, t1_bc_h, t1_Wl_z, t1_bl_z,
                                     t1_Wl_h, t1_bl_h, att1, hbuf);
  // 7. VAE head -> mu, logvar, d
  k_vae<<<NN, 128, 0, stream>>>(hbuf, W_mu, b_mu, W_lv, b_lv, W_dec, b_dec,
                                mu_o, lv_o, dvec);
  // 8. decoder features -> MdecT hi/lo
  k_dec_feat<<<NN, 128, 0, stream>>>(dvec, td_Wc_z, td_Wc_h, MDTh, MDTl);
  // 9. G2 = AnT @ Mdec
  k_gemm_split<<<dim3(128 / 64, NN / 64), 128, 0, stream>>>(Ahi, Alo, MDTh, MDTl, G2, 128);
  // 10. decoder gate -> recon_x
  k_dec_gate<<<NN, 64, 0, stream>>>(G2, td_bc_z, td_bc_h, td_Wl_z, td_bl_z,
                                    td_Wl_h, td_bl_h, recon);
}
